// xLSTMOperation_80023830659665
// MI455X (gfx1250) — matrix operand order repaired and checked by reading, not compiled, not run on hardware
//
#include <hip/hip_runtime.h>
#include <hip/hip_bf16.h>
#include <stdint.h>

typedef _Float16 h16;
typedef __attribute__((ext_vector_type(16))) _Float16 v16h;
typedef __attribute__((ext_vector_type(8)))  _Float16 v8h;
typedef __attribute__((ext_vector_type(8)))  float    v8f;

// Problem constants
#define BB     16
#define CC     192
#define HH_    32
#define WW_    32
#define SS     1024
#define INNER  384
#define NHEAD  12
#define DHEAD  32
#define ROWS   (BB * SS)      // 16384
#define PADW   34             // 32 + 2 halo for 3x3 conv

__device__ __forceinline__ float siluf(float x) { return x / (1.f + __expf(-x)); }

// ---------------------------------------------------------------------------
// Weight conversion: f32 [K][N] -> f16 transposed [N][K] (so B-fragments of a
// WMMA 16x16x32 tile are two contiguous 16B loads per lane).
// ---------------------------------------------------------------------------
__global__ void k_transpose_w(const float* __restrict__ src, h16* __restrict__ dst,
                              int K, int N) {
  int idx = blockIdx.x * blockDim.x + threadIdx.x;
  if (idx >= K * N) return;
  int n = idx / K, k = idx % K;
  dst[idx] = (h16)src[k * N + n];
}

// conv2_w [CO][CI][3][3] f32 -> w9 [tap][CO][CI] f16 (B^T layout per tap)
__global__ void k_conv_w(const float* __restrict__ src, h16* __restrict__ dst) {
  int idx = blockIdx.x * blockDim.x + threadIdx.x;
  if (idx >= 9 * CC * CC) return;
  int tap = idx / (CC * CC);
  int rem = idx - tap * CC * CC;
  int co = rem / CC, ci = rem % CC;
  dst[idx] = (h16)src[(co * CC + ci) * 9 + tap];
}

// ---------------------------------------------------------------------------
// LayerNorm + NCHW->NSC transpose. One wave per row (192 ch = 6/lane).
// Emits: seqf (f32 raw), seq16 (f16 raw), hn16 (f16 normalized).
// ---------------------------------------------------------------------------
__global__ void k_ln(const float* __restrict__ x, const float* __restrict__ lw,
                     const float* __restrict__ lb, float* __restrict__ seqf,
                     h16* __restrict__ seq16, h16* __restrict__ hn16) {
  int lane = threadIdx.x & 31;
  int wv   = threadIdx.x >> 5;
  size_t row = (size_t)blockIdx.x * 8 + wv;
  int b = (int)(row >> 10), p = (int)(row & 1023);
  float vals[6], s = 0.f, s2 = 0.f;
#pragma unroll
  for (int j = 0; j < 6; ++j) {
    int c = lane + j * 32;
    float v = x[((size_t)b * CC + c) * SS + p];
    vals[j] = v; s += v; s2 += v * v;
  }
#pragma unroll
  for (int off = 16; off > 0; off >>= 1) { s += __shfl_xor(s, off); s2 += __shfl_xor(s2, off); }
  float mu = s * (1.f / 192.f);
  float var = s2 * (1.f / 192.f) - mu * mu;
  float rs = rsqrtf(var + 1e-5f);
#pragma unroll
  for (int j = 0; j < 6; ++j) {
    int c = lane + j * 32;
    float v = vals[j];
    seqf [row * CC + c] = v;
    seq16[row * CC + c] = (h16)v;
    hn16 [row * CC + c] = (h16)((v - mu) * rs * lw[c] + lb[c]);
  }
}

// ---------------------------------------------------------------------------
// WMMA fragment helpers (A row-major [16,K] f16; B^T [N][K] f16).
// ---------------------------------------------------------------------------
__device__ __forceinline__ v16h frag_a(const h16* __restrict__ A, int lda,
                                       int k0, int lane) {
  int row = lane & 15, hi = lane >> 4;
  const h16* p = A + (size_t)row * lda + k0 + hi * 16;
  v8h a0 = *(const v8h*)p;
  v8h a1 = *(const v8h*)(p + 8);
  v16h r;
#pragma unroll
  for (int j = 0; j < 8; ++j) { r[j] = a0[j]; r[8 + j] = a1[j]; }
  return r;
}
__device__ __forceinline__ v16h frag_b(const h16* __restrict__ Bt, int K,
                                       int k0, int lane) {
  int col = lane & 15, hi = lane >> 4;
  const h16* p = Bt + (size_t)col * K + k0 + hi * 16;
  v8h b0 = *(const v8h*)p;
  v8h b1 = *(const v8h*)(p + 8);
  v16h r;
#pragma unroll
  for (int j = 0; j < 8; ++j) { r[j] = b0[j]; r[8 + j] = b1[j]; }
  return r;
}

// ---------------------------------------------------------------------------
// Generic WMMA GEMM: C[M,N] = A[M,K] * B^T[N,K] + bias.
// Block = 128 threads = 4 waves; wave computes a 16x16 tile; grid(N/64, M/16).
// ---------------------------------------------------------------------------
__global__ void k_gemm(const h16* __restrict__ A, const h16* __restrict__ Bt,
                       float* __restrict__ Cout, const float* __restrict__ bias,
                       int N, int K, int ldc) {
  int lane  = threadIdx.x & 31;
  int wave  = threadIdx.x >> 5;
  int ntile = blockIdx.x * 4 + wave;
  int mtile = blockIdx.y;
  const h16* Arow = A  + (size_t)mtile * 16 * K;
  const h16* Bcol = Bt + (size_t)ntile * 16 * K;
  v8f acc = {};
  for (int k0 = 0; k0 < K; k0 += 32) {
    __builtin_prefetch(Arow + ((size_t)(lane & 15)) * K + k0 + 32, 0, 3);
    v16h av = frag_a(Arow, K, k0, lane);
    v16h bv = frag_b(Bcol, K, k0, lane);
    acc = __builtin_amdgcn_wmma_f32_16x16x32_f16(false, av, false, bv,
                                                 (short)0, acc, false, false);
  }
  int col = lane & 15, rbase = (lane >> 4) * 8;
#pragma unroll
  for (int r = 0; r < 8; ++r) {
    int gr = mtile * 16 + rbase + r;
    int gc = ntile * 16 + col;
    float v = acc[r] + (bias ? bias[gc] : 0.f);
    Cout[(size_t)gr * ldc + gc] = v;
  }
}

// ---------------------------------------------------------------------------
// Causal depthwise conv1d (k=4) + SiLU over x_in (= up[:, :384]).
// ---------------------------------------------------------------------------
__global__ void k_conv1d(const float* __restrict__ up, const float* __restrict__ ck,
                         const float* __restrict__ cb, float* __restrict__ xact,
                         h16* __restrict__ xact16) {
  size_t idx = (size_t)blockIdx.x * blockDim.x + threadIdx.x;
  if (idx >= (size_t)ROWS * INNER) return;
  int ch = (int)(idx % INNER);
  size_t row = idx / INNER;
  int b = (int)(row >> 10), s = (int)(row & 1023);
  float acc = cb[ch];
#pragma unroll
  for (int t = 0; t < 4; ++t) {
    int s2 = s - 3 + t;
    if (s2 >= 0) acc += up[((size_t)(b << 10) + s2) * 768 + ch] * ck[ch * 4 + t];
  }
  float sv = siluf(acc);
  xact[idx] = sv;
  xact16[idx] = (h16)sv;
}

// ---------------------------------------------------------------------------
// Gate pre-activations: i/f = [q,k,v] @ w_{i,f} + b (NH=12 wide -> dot prods).
// ---------------------------------------------------------------------------
__global__ void k_gates(const float* __restrict__ q, const float* __restrict__ k,
                        const float* __restrict__ up, const float* __restrict__ wi,
                        const float* __restrict__ bi, const float* __restrict__ wf,
                        const float* __restrict__ bf, float* __restrict__ ipre,
                        float* __restrict__ fpre) {
  int idx = blockIdx.x * blockDim.x + threadIdx.x;
  if (idx >= ROWS * NHEAD) return;
  int hh = idx % NHEAD;
  size_t row = idx / NHEAD;
  float ai = bi[hh], af = bf[hh];
  for (int j = 0; j < INNER; ++j) {
    float qv = q[row * INNER + j];
    ai += qv * wi[j * NHEAD + hh]; af += qv * wf[j * NHEAD + hh];
  }
  for (int j = 0; j < INNER; ++j) {
    float kv = k[row * INNER + j];
    ai += kv * wi[(INNER + j) * NHEAD + hh]; af += kv * wf[(INNER + j) * NHEAD + hh];
  }
  for (int j = 0; j < INNER; ++j) {
    float vv = up[row * 768 + j];
    ai += vv * wi[(2 * INNER + j) * NHEAD + hh]; af += vv * wf[(2 * INNER + j) * NHEAD + hh];
  }
  ipre[idx] = ai; fpre[idx] = af;
}

// ---------------------------------------------------------------------------
// Sequential mLSTM scan. One wave32 per (batch, head); lane d owns row d of
// the 32x32 matrix memory (32 VGPRs of state). q/k broadcast via readlane.
// ---------------------------------------------------------------------------
__global__ void k_scan(const float* __restrict__ q, const float* __restrict__ k,
                       const float* __restrict__ up, const float* __restrict__ ipre,
                       const float* __restrict__ fpre, float* __restrict__ hs) {
  int lane = threadIdx.x;
  int bh = blockIdx.x;              // 0..191
  int b = bh / NHEAD, hh = bh % NHEAD;
  float Cm[32];
#pragma unroll
  for (int e = 0; e < 32; ++e) Cm[e] = 0.f;
  float nst = 0.f, mst = 0.f;
  const float kscale = 0.17677669529663687f;   // 1/sqrt(32)
  for (int t = 0; t < SS; ++t) {
    size_t row = (size_t)(b << 10) + t;
    float qd = q[row * INNER + hh * DHEAD + lane];
    float kd = k[row * INNER + hh * DHEAD + lane] * kscale;
    float vd = up[row * 768 + hh * DHEAD + lane];
    float it = ipre[row * NHEAD + hh];
    float ft = fpre[row * NHEAD + hh];
    float lf = fminf(ft, 0.f) - __logf(1.f + __expf(-fabsf(ft)));  // log_sigmoid
    float mnew = fmaxf(lf + mst, it);
    float fs = __expf(lf + mst - mnew);
    float is = __expf(it - mnew);
    nst = fs * nst + is * kd;
    float dot = nst * qd;
#pragma unroll
    for (int off = 16; off > 0; off >>= 1) dot += __shfl_xor(dot, off);
    float denom = fmaxf(fabsf(dot), 1.f);
    float hd = 0.f;
#pragma unroll
    for (int e = 0; e < 32; ++e) {
      float ke = __shfl(kd, e);
      float qe = __shfl(qd, e);
      Cm[e] = fs * Cm[e] + is * vd * ke;   // rank-1 update
      hd = fmaf(Cm[e], qe, hd);            // fused matvec
    }
    hs[row * INNER + hh * DHEAD + lane] = hd / denom;
    mst = mnew;
  }
}

// ---------------------------------------------------------------------------
// Per-head groupnorm + skip/gating; one wave per (row, head).
// ---------------------------------------------------------------------------
__global__ void k_gn(const float* __restrict__ hsb, const float* __restrict__ xact,
                     const float* __restrict__ up, const float* __restrict__ mhw,
                     const float* __restrict__ mhb, const float* __restrict__ skip,
                     h16* __restrict__ hout) {
  int lane = threadIdx.x & 31;
  int head = threadIdx.x >> 5;
  size_t row = blockIdx.x;
  int c = head * DHEAD + lane;
  float h = hsb[row * INNER + c];
  float s = h, s2 = h * h;
#pragma unroll
  for (int off = 16; off > 0; off >>= 1) { s += __shfl_xor(s, off); s2 += __shfl_xor(s2, off); }
  float mu = s * (1.f / 32.f);
  float var = s2 * (1.f / 32.f) - mu * mu;
  float hn = (h - mu) * rsqrtf(var + 1e-5f) * mhw[c] + mhb[c];
  float xa = xact[row * INNER + c];
  float z  = up[row * 768 + INNER + c];
  hout[row * INNER + c] = (h16)((hn + skip[c] * xa) * siluf(z));
}

// ---------------------------------------------------------------------------
// Zero the padded NHWC f16 conv-input buffer (halo must be zero).
// ---------------------------------------------------------------------------
__global__ void k_zero16(h16* __restrict__ p, size_t n) {
  size_t i = (size_t)blockIdx.x * blockDim.x + threadIdx.x;
  if (i < n) p[i] = (h16)0.f;
}

// ---------------------------------------------------------------------------
// Fused down-proj + linear-branch WMMA GEMMs; epilogue builds
// y = seq + (hout@w_down + b_down) + silu(seq@w_lin + b_lin) and writes it
// into padded NHWC f16 for the 3x3 conv.
// ---------------------------------------------------------------------------
__global__ void k_down_lin(const h16* __restrict__ hout, const h16* __restrict__ wtd,
                           const h16* __restrict__ seq16, const h16* __restrict__ wtl,
                           const float* __restrict__ seqf, const float* __restrict__ bd,
                           const float* __restrict__ bl, h16* __restrict__ ypad) {
  int lane  = threadIdx.x & 31;
  int wave  = threadIdx.x >> 5;
  int ntile = blockIdx.x * 4 + wave;     // 0..11 (192 cols)
  int mtile = blockIdx.y;                // 0..1023
  v8f acc1 = {}, acc2 = {};
  {
    const h16* Arow = hout + (size_t)mtile * 16 * INNER;
    const h16* Bcol = wtd  + (size_t)ntile * 16 * INNER;
    for (int k0 = 0; k0 < INNER; k0 += 32) {
      v16h av = frag_a(Arow, INNER, k0, lane);
      v16h bv = frag_b(Bcol, INNER, k0, lane);
      acc1 = __builtin_amdgcn_wmma_f32_16x16x32_f16(false, av, false, bv,
                                                    (short)0, acc1, false, false);
    }
  }
  {
    const h16* Arow = seq16 + (size_t)mtile * 16 * CC;
    const h16* Bcol = wtl   + (size_t)ntile * 16 * CC;
    for (int k0 = 0; k0 < CC; k0 += 32) {
      v16h av = frag_a(Arow, CC, k0, lane);
      v16h bv = frag_b(Bcol, CC, k0, lane);
      acc2 = __builtin_amdgcn_wmma_f32_16x16x32_f16(false, av, false, bv,
                                                    (short)0, acc2, false, false);
    }
  }
  int col = lane & 15, rbase = (lane >> 4) * 8;
#pragma unroll
  for (int r = 0; r < 8; ++r) {
    int gr = mtile * 16 + rbase + r;
    int gc = ntile * 16 + col;
    float x1 = seqf[(size_t)gr * CC + gc] + acc1[r] + bd[gc];
    float x2 = siluf(acc2[r] + bl[gc]);
    float y = x1 + x2;
    int b = gr >> 10, p = gr & 1023;
    int ph = p >> 5, pw = p & 31;
    ypad[(((size_t)(b * PADW + 1 + ph)) * PADW + 1 + pw) * CC + gc] = (h16)y;
  }
}

// ---------------------------------------------------------------------------
// 3x3 conv as 9 shifted WMMA GEMMs over padded NHWC f16.
// The 16-pixel A tile per tap is CONTIGUOUS (6144B) and shared by all 4 waves,
// so it is staged block-cooperatively into LDS with the CDNA5 async copy path
// (global_load_async_to_lds_b128, ASYNCcnt), double-buffered across taps.
// ---------------------------------------------------------------------------
__device__ __forceinline__ void wait_asynccnt_3() {
#if __has_builtin(__builtin_amdgcn_s_wait_asynccnt)
  __builtin_amdgcn_s_wait_asynccnt(3);
#else
  asm volatile("s_wait_asynccnt 0x3" ::: "memory");
#endif
}
__device__ __forceinline__ void wait_asynccnt_0() {
#if __has_builtin(__builtin_amdgcn_s_wait_asynccnt)
  __builtin_amdgcn_s_wait_asynccnt(0);
#else
  asm volatile("s_wait_asynccnt 0x0" ::: "memory");
#endif
}

__global__ void k_conv2(const h16* __restrict__ ypad, const h16* __restrict__ w9,
                        const float* __restrict__ cb, float* __restrict__ yconv) {
  __shared__ __align__(16) h16 smemA[2][16 * CC];   // 2 x 6144B double buffer
  int lane  = threadIdx.x & 31;
  int wave  = threadIdx.x >> 5;
  int ntile = blockIdx.x * 4 + wave;     // 0..11
  int tid   = blockIdx.y;                // b*64 + h*2 + whalf
  int b = tid >> 6, h = (tid >> 1) & 31, w0 = (tid & 1) * 16;

  uint32_t ldsbase = (uint32_t)(uintptr_t)(void*)&smemA[0][0];

  // Stage one tap's 16x192 f16 A tile (contiguous) into LDS buffer `buf`.
  // 384 x 16B chunks, split over 128 threads: 3 async instrs per wave.
  auto stage = [&](int tap, int buf) {
    int kh = tap / 3, kw = tap % 3;
    const h16* src = ypad + (((size_t)(b * PADW + h + kh)) * PADW + (w0 + kw)) * CC;
    uint32_t dstbase = ldsbase + (uint32_t)buf * (16 * CC * 2);
#pragma unroll
    for (int j = 0; j < 3; ++j) {
      uint32_t c16 = (uint32_t)threadIdx.x + (uint32_t)j * 128;  // chunk id
      uint64_t ga = (uint64_t)(uintptr_t)src + (uint64_t)c16 * 16;
      uint32_t la = dstbase + c16 * 16;
      asm volatile("global_load_async_to_lds_b128 %0, %1, off"
                   :: "v"(la), "v"(ga) : "memory");
    }
  };

  stage(0, 0);
  v8f acc = {};
#pragma unroll
  for (int tap = 0; tap < 9; ++tap) {
    if (tap < 8) { stage(tap + 1, (tap + 1) & 1); wait_asynccnt_3(); }
    else         { wait_asynccnt_0(); }
    __syncthreads();                       // publish all waves' staged chunks
    const h16* Asm  = &smemA[tap & 1][0];  // ds_load_b128 fragment reads
    const h16* Bcol = w9 + (size_t)tap * CC * CC + (size_t)ntile * 16 * CC;
    for (int k0 = 0; k0 < CC; k0 += 32) {
      v16h av = frag_a(Asm, CC, k0, lane);
      v16h bv = frag_b(Bcol, CC, k0, lane);
      acc = __builtin_amdgcn_wmma_f32_16x16x32_f16(false, av, false, bv,
                                                   (short)0, acc, false, false);
    }
    __syncthreads();                       // buffer reuse hazard fence
  }
  int col = lane & 15, rbase = (lane >> 4) * 8;
#pragma unroll
  for (int r = 0; r < 8; ++r) {
    int pix = h * 32 + w0 + rbase + r;
    int gc = ntile * 16 + col;
    yconv[((size_t)(b << 10) + pix) * CC + gc] = acc[r] + cb[gc];
  }
}

// ---------------------------------------------------------------------------
// BatchNorm stats (training mode, biased var): one wave per channel.
// ---------------------------------------------------------------------------
__global__ void k_bnstats(const float* __restrict__ yc, const float* __restrict__ g,
                          const float* __restrict__ bb, float* __restrict__ ab) {
  int c = blockIdx.x, lane = threadIdx.x;
  float s = 0.f, s2 = 0.f;
  for (int i = lane; i < ROWS; i += 32) {
    float v = yc[(size_t)i * CC + c];
    s += v; s2 += v * v;
  }
#pragma unroll
  for (int off = 16; off > 0; off >>= 1) { s += __shfl_xor(s, off); s2 += __shfl_xor(s2, off); }
  if (lane == 0) {
    float mu = s * (1.f / (float)ROWS);
    float var = s2 * (1.f / (float)ROWS) - mu * mu;
    float sc = g[c] * rsqrtf(var + 1e-5f);
    ab[c * 2] = sc;
    ab[c * 2 + 1] = bb[c] - mu * sc;
  }
}

// BN apply + LeakyReLU + NSC -> NCHW transpose into d_out.
__global__ void k_bnout(const float* __restrict__ yc, const float* __restrict__ ab,
                        float* __restrict__ out) {
  size_t idx = (size_t)blockIdx.x * blockDim.x + threadIdx.x;
  if (idx >= (size_t)BB * CC * SS) return;
  int p = (int)(idx & 1023);
  int c = (int)((idx >> 10) % CC);
  int b = (int)(idx / ((size_t)CC * SS));
  float v = yc[((size_t)(b << 10) + p) * CC + c];
  v = v * ab[c * 2] + ab[c * 2 + 1];
  out[idx] = (v >= 0.f) ? v : 0.01f * v;
}

// ---------------------------------------------------------------------------
// Launcher
// ---------------------------------------------------------------------------
extern "C" void kernel_launch(void* const* d_in, const int* in_sizes, int n_in,
                              void* d_out, int out_size, void* d_ws, size_t ws_size,
                              hipStream_t stream) {
  const float* x      = (const float*)d_in[0];
  const float* ln_w   = (const float*)d_in[1];
  const float* ln_b   = (const float*)d_in[2];
  const float* w_up   = (const float*)d_in[3];
  const float* b_up   = (const float*)d_in[4];
  const float* conv_k = (const float*)d_in[5];
  const float* conv_b = (const float*)d_in[6];
  const float* w_q    = (const float*)d_in[7];
  const float* w_k    = (const float*)d_in[8];
  const float* w_i    = (const float*)d_in[9];
  const float* b_i    = (const float*)d_in[10];
  const float* w_f    = (const float*)d_in[11];
  const float* b_f    = (const float*)d_in[12];
  const float* skip   = (const float*)d_in[13];
  const float* mh_w   = (const float*)d_in[14];
  const float* mh_b   = (const float*)d_in[15];
  const float* w_down = (const float*)d_in[16];
  const float* b_down = (const float*)d_in[17];
  const float* w_lin  = (const float*)d_in[18];
  const float* b_lin  = (const float*)d_in[19];
  const float* conv2w = (const float*)d_in[20];
  const float* conv2b = (const float*)d_in[21];
  const float* bn_g   = (const float*)d_in[22];
  const float* bn_b   = (const float*)d_in[23];
  float* out = (float*)d_out;

  // Workspace carve-out (aligned to 256B)
  char* base = (char*)d_ws;
  size_t cur = 0;
  auto alloc = [&](size_t bytes) -> char* {
    char* p = base + cur;
    cur = (cur + bytes + 255) & ~(size_t)255;
    return p;
  };
  float* seqf  = (float*)alloc((size_t)ROWS * CC * 4);
  h16*   seq16 = (h16*)  alloc((size_t)ROWS * CC * 2);
  h16*   hn16  = (h16*)  alloc((size_t)ROWS * CC * 2);
  h16*   wtup  = (h16*)  alloc((size_t)768 * CC * 2);
  h16*   wtq   = (h16*)  alloc((size_t)INNER * INNER * 2);
  h16*   wtk   = (h16*)  alloc((size_t)INNER * INNER * 2);
  h16*   wtd   = (h16*)  alloc((size_t)CC * INNER * 2);
  h16*   wtl   = (h16*)  alloc((size_t)CC * CC * 2);
  h16*   w9    = (h16*)  alloc((size_t)9 * CC * CC * 2);
  float* up    = (float*)alloc((size_t)ROWS * 768 * 4);   // [x_in | z]
  float* xact  = (float*)alloc((size_t)ROWS * INNER * 4);
  h16*   xact16= (h16*)  alloc((size_t)ROWS * INNER * 2);
  float* qbuf  = (float*)alloc((size_t)ROWS * INNER * 4);
  float* kbuf  = (float*)alloc((size_t)ROWS * INNER * 4);
  float* ipre  = (float*)alloc((size_t)ROWS * NHEAD * 4);
  float* fpre  = (float*)alloc((size_t)ROWS * NHEAD * 4);
  float* hsbuf = (float*)alloc((size_t)ROWS * INNER * 4);
  h16*   hout  = (h16*)  alloc((size_t)ROWS * INNER * 2);
  h16*   ypad  = (h16*)  alloc((size_t)BB * PADW * PADW * CC * 2);
  float* yconv = (float*)alloc((size_t)ROWS * CC * 4);
  float* bnab  = (float*)alloc((size_t)CC * 2 * 4);

  // 1) Weight conversion / transposition
  k_transpose_w<<<(CC * 768 + 255) / 256, 256, 0, stream>>>(w_up, wtup, CC, 768);
  k_transpose_w<<<(INNER * INNER + 255) / 256, 256, 0, stream>>>(w_q, wtq, INNER, INNER);
  k_transpose_w<<<(INNER * INNER + 255) / 256, 256, 0, stream>>>(w_k, wtk, INNER, INNER);
  k_transpose_w<<<(INNER * CC + 255) / 256, 256, 0, stream>>>(w_down, wtd, INNER, CC);
  k_transpose_w<<<(CC * CC + 255) / 256, 256, 0, stream>>>(w_lin, wtl, CC, CC);
  k_conv_w<<<(9 * CC * CC + 255) / 256, 256, 0, stream>>>(conv2w, w9);

  // 2) LayerNorm + transpose
  k_ln<<<ROWS / 8, 256, 0, stream>>>(x, ln_w, ln_b, seqf, seq16, hn16);

  // 3) Up-projection GEMM: [16384,192] x [192,768]
  k_gemm<<<dim3(768 / 64, ROWS / 16), 128, 0, stream>>>(hn16, wtup, up, b_up, 768, CC, 768);

  // 4) Causal depthwise conv1d + SiLU
  k_conv1d<<<(ROWS * INNER + 255) / 256, 256, 0, stream>>>(up, conv_k, conv_b, xact, xact16);

  // 5) q/k GEMMs: [16384,384] x [384,384]
  k_gemm<<<dim3(INNER / 64, ROWS / 16), 128, 0, stream>>>(xact16, wtq, qbuf, nullptr, INNER, INNER, INNER);
  k_gemm<<<dim3(INNER / 64, ROWS / 16), 128, 0, stream>>>(xact16, wtk, kbuf, nullptr, INNER, INNER, INNER);

  // 6) Gate pre-activations
  k_gates<<<(ROWS * NHEAD + 127) / 128, 128, 0, stream>>>(qbuf, kbuf, up, w_i, b_i, w_f, b_f, ipre, fpre);

  // 7) mLSTM recurrence: one wave32 per (batch, head)
  k_scan<<<BB * NHEAD, 32, 0, stream>>>(qbuf, kbuf, up, ipre, fpre, hsbuf);

  // 8) Per-head groupnorm + skip + z-gate
  k_gn<<<ROWS, INNER, 0, stream>>>(hsbuf, xact, up, mh_w, mh_b, skip, hout);

  // 9) Zero padded conv input, then fused down+lin GEMM epilogue into it
  {
    size_t n = (size_t)BB * PADW * PADW * CC;
    k_zero16<<<(unsigned)((n + 255) / 256), 256, 0, stream>>>(ypad, n);
  }
  k_down_lin<<<dim3(CC / 64, ROWS / 16), 128, 0, stream>>>(hout, wtd, seq16, wtl,
                                                           seqf, b_down, b_lin, ypad);

  // 10) 3x3 conv: async-LDS staged A, 9 shifted WMMA GEMMs
  k_conv2<<<dim3(CC / 64, BB * HH_ * 2), 128, 0, stream>>>(ypad, w9, conv2b, yconv);

  // 11) BatchNorm stats + apply + LeakyReLU
  k_bnstats<<<CC, 32, 0, stream>>>(yconv, bn_g, bn_b, bnab);
  k_bnout<<<(unsigned)(((size_t)BB * CC * SS + 255) / 256), 256, 0, stream>>>(yconv, bnab, out);

  (void)in_sizes; (void)n_in; (void)out_size; (void)ws_size;
}
